// RoutingCritic_71579924955603
// MI455X (gfx1250) — matrix operand order repaired and checked by reading, not compiled, not run on hardware
//
#include <hip/hip_runtime.h>
#include <math.h>

typedef __attribute__((ext_vector_type(16))) __bf16 v16bf;
typedef __attribute__((ext_vector_type(8)))  float  v8f;
typedef __attribute__((ext_vector_type(8)))  unsigned int v8u;

__device__ __forceinline__ unsigned short f2bf(float f) {
    unsigned u = __float_as_uint(f);
    unsigned r = (u + 0x7FFFu + ((u >> 16) & 1u)) >> 16;
    return (unsigned short)r;
}

__device__ __forceinline__ v16bf mkfrag(uint4 a, uint4 b) {
    v8u u;
    u[0] = a.x; u[1] = a.y; u[2] = a.z; u[3] = a.w;
    u[4] = b.x; u[5] = b.y; u[6] = b.z; u[7] = b.w;
    return __builtin_bit_cast(v16bf, u);
}

__device__ __forceinline__ v8f wmma_bf16(v16bf a, v16bf b, v8f c) {
    return __builtin_amdgcn_wmma_f32_16x16x32_bf16(false, a, false, b, (short)0, c, false, false);
}

// ---------------------------------------------------------------------------
// Pack a [K,128] f32 weight into bf16 B-fragment-major layout:
// out[((kc*8+nt)*32+lane)*16 + e], lane: col = nt*16+(lane&15), k = kc*32+(lane>>4)*8+(e&7)+(e>>3)*16
// k >= Kact is zero-padded.
// ---------------------------------------------------------------------------
__global__ void pack_b_frags(const float* __restrict__ w, unsigned short* __restrict__ out,
                             int Kact, int KC) {
    int gid = blockIdx.x * 256 + threadIdx.x;
    int total = KC * 8 * 32;
    if (gid >= total) return;
    int lane = gid & 31;
    int nt   = (gid >> 5) & 7;
    int kc   = gid >> 8;
    int n    = nt * 16 + (lane & 15);
    unsigned short* o = out + (size_t)gid * 16;
#pragma unroll
    for (int e = 0; e < 16; ++e) {
        int k = kc * 32 + (lane >> 4) * 8 + (e & 7) + (e >> 3) * 16;
        float v = (k < Kact) ? w[(size_t)k * 128 + n] : 0.f;
        o[e] = f2bf(v);
    }
}

// ---------------------------------------------------------------------------
// Encoder GEMM: [nrows,KDIM] f32 -> [nrows,128] (f32 and/or bf16)
// 64 rows / block, 4 waves, 1 M-tile per wave.
// ---------------------------------------------------------------------------
template <int KDIM, int KC, int STR>
__global__ __launch_bounds__(128) void encoder_kernel(
    const float* __restrict__ feat, const unsigned short* __restrict__ wfrag,
    const float* __restrict__ bias, float* __restrict__ outf,
    unsigned short* __restrict__ outh, int nrows) {
    extern __shared__ char smem[];
    unsigned short* Ain = (unsigned short*)smem;  // [64][STR]
    const int tid = threadIdx.x;
    const int n0 = blockIdx.x * 64;

    for (int idx = tid; idx < 64 * KDIM; idx += 128) {
        int row = idx / KDIM, c = idx % KDIM;
        int rr = n0 + row; if (rr >= nrows) rr = nrows - 1;
        Ain[row * STR + c] = f2bf(feat[(size_t)rr * KDIM + c]);
    }
    __syncthreads();

    const int lane = tid & 31, wv = tid >> 5;
    const int lo = lane & 15, hi = lane >> 4;
    const unsigned short* Arow = Ain + (wv * 16 + lo) * STR;
    const uint4* wq = (const uint4*)wfrag;

    for (int nt = 0; nt < 8; ++nt) {
        float bv = bias[nt * 16 + lo];
        v8f acc = {bv, bv, bv, bv, bv, bv, bv, bv};
#pragma unroll
        for (int kc = 0; kc < KC; ++kc) {
            int kb = kc * 32 + hi * 8;
            uint4 a0 = *(const uint4*)(Arow + kb);
            uint4 a1 = *(const uint4*)(Arow + kb + 16);
            const uint4* bp = wq + ((((kc * 8 + nt) * 32) + lane) << 1);
            acc = wmma_bf16(mkfrag(a0, a1), mkfrag(bp[0], bp[1]), acc);
        }
#pragma unroll
        for (int r = 0; r < 8; ++r) {
            int row = wv * 16 + r + 8 * hi;
            int node = n0 + row;
            if (node < nrows) {
                size_t o = (size_t)node * 128 + nt * 16 + lo;
                if (outf) outf[o] = acc[r];
                if (outh) outh[o] = f2bf(acc[r]);
            }
        }
    }
}

// ---------------------------------------------------------------------------
// Message MLP + scatter-add.  64 edges / block, 4 waves.
// m_in = [x[src] | x[dst] | ea | cong | 0-pad]  (K padded 385 -> 416 = 13*32)
// ---------------------------------------------------------------------------
__global__ __launch_bounds__(128) void msg_kernel(
    const unsigned short* __restrict__ xh, const unsigned short* __restrict__ ea,
    const float* __restrict__ cong, const int* __restrict__ src,
    const int* __restrict__ dst, const unsigned short* __restrict__ w1f,
    const float* __restrict__ b1, const unsigned short* __restrict__ w2f,
    const float* __restrict__ b2, float* __restrict__ agg, int E_) {
    extern __shared__ char smem[];
    unsigned short* Ain = (unsigned short*)smem;             // [64][424]
    unsigned short* Hs  = (unsigned short*)(smem + 54272);   // [64][136]
    int* s_src = (int*)(smem + 71680);
    int* s_dst = (int*)(smem + 71936);

    const int tid = threadIdx.x;
    const int e0 = blockIdx.x * 64;

    if (tid < 64) {
        int e = e0 + tid; if (e >= E_) e = E_ - 1;
        s_src[tid] = src[e];
        s_dst[tid] = dst[e];
    }
    __syncthreads();

    const unsigned int* xh32 = (const unsigned int*)xh;  // 64 dwords / row
    const unsigned int* ea32 = (const unsigned int*)ea;
    for (int idx = tid; idx < 64 * 64; idx += 128) {
        int row = idx >> 6, c = idx & 63;
        int e = e0 + row; if (e >= E_) e = E_ - 1;
        unsigned int* arow = (unsigned int*)(Ain + row * 424);
        arow[c]       = xh32[(size_t)s_src[row] * 64 + c];   // cols   0..127
        arow[64 + c]  = xh32[(size_t)s_dst[row] * 64 + c];   // cols 128..255
        arow[128 + c] = ea32[(size_t)e * 64 + c];            // cols 256..383
    }
    for (int idx = tid; idx < 64 * 16; idx += 128) {
        int row = idx >> 4, c = idx & 15;
        unsigned int v = 0;
        if (c == 0) v = (unsigned int)f2bf(cong[s_src[row]]);  // col 384, rest 0
        ((unsigned int*)(Ain + row * 424))[192 + c] = v;       // cols 384..415
    }
    __syncthreads();

    const int lane = tid & 31, wv = tid >> 5;
    const int lo = lane & 15, hi = lane >> 4;
    const unsigned short* Arow = Ain + (wv * 16 + lo) * 424;
    const uint4* w1q = (const uint4*)w1f;
    const uint4* w2q = (const uint4*)w2f;

    // GEMM1: [64,416] x [416,128] + ReLU -> Hs bf16
    for (int nt = 0; nt < 8; ++nt) {
        float bv = b1[nt * 16 + lo];
        v8f acc = {bv, bv, bv, bv, bv, bv, bv, bv};
#pragma unroll
        for (int kc = 0; kc < 13; ++kc) {
            int kb = kc * 32 + hi * 8;
            uint4 a0 = *(const uint4*)(Arow + kb);
            uint4 a1 = *(const uint4*)(Arow + kb + 16);
            const uint4* bp = w1q + ((((kc * 8 + nt) * 32) + lane) << 1);
            if (kc < 12) __builtin_prefetch((const void*)(bp + 512), 0, 1);
            acc = wmma_bf16(mkfrag(a0, a1), mkfrag(bp[0], bp[1]), acc);
        }
#pragma unroll
        for (int r = 0; r < 8; ++r) {
            float v = fmaxf(acc[r], 0.f);
            Hs[(wv * 16 + r + 8 * hi) * 136 + nt * 16 + lo] = f2bf(v);
        }
    }
    __syncthreads();

    // GEMM2: [64,128] x [128,128] -> atomic scatter into agg[dst]
    const unsigned short* Hrow = Hs + (wv * 16 + lo) * 136;
    for (int nt = 0; nt < 8; ++nt) {
        float bv = b2[nt * 16 + lo];
        v8f acc = {bv, bv, bv, bv, bv, bv, bv, bv};
#pragma unroll
        for (int kc = 0; kc < 4; ++kc) {
            int kb = kc * 32 + hi * 8;
            uint4 a0 = *(const uint4*)(Hrow + kb);
            uint4 a1 = *(const uint4*)(Hrow + kb + 16);
            const uint4* bp = w2q + ((((kc * 8 + nt) * 32) + lane) << 1);
            acc = wmma_bf16(mkfrag(a0, a1), mkfrag(bp[0], bp[1]), acc);
        }
#pragma unroll
        for (int r = 0; r < 8; ++r) {
            int row = wv * 16 + r + 8 * hi;
            int e = e0 + row;
            if (e < E_)
                unsafeAtomicAdd(&agg[(size_t)s_dst[row] * 128 + nt * 16 + lo], acc[r]);
        }
    }
}

// ---------------------------------------------------------------------------
// Node update: u_in=[x|agg] (K=256), MLP, residual, LayerNorm. In-place on x.
// ---------------------------------------------------------------------------
__global__ __launch_bounds__(128) void upd_kernel(
    float* __restrict__ x, const float* __restrict__ aggv,
    const unsigned short* __restrict__ w1f, const float* __restrict__ b1,
    const unsigned short* __restrict__ w2f, const float* __restrict__ b2,
    const float* __restrict__ lnw, const float* __restrict__ lnb,
    unsigned short* __restrict__ xh, int n) {
    extern __shared__ char smem[];
    unsigned short* Ain = (unsigned short*)smem;             // [64][264]
    unsigned short* Hs  = (unsigned short*)(smem + 33792);   // [64][136]
    float* Rs           = (float*)(smem + 51200);            // [64][132]

    const int tid = threadIdx.x;
    const int n0 = blockIdx.x * 64;

    for (int idx = tid; idx < 64 * 128; idx += 128) {
        int row = idx >> 7, c = idx & 127;
        int node = n0 + row; if (node >= n) node = n - 1;
        Ain[row * 264 + c]       = f2bf(x[(size_t)node * 128 + c]);
        Ain[row * 264 + 128 + c] = f2bf(aggv[(size_t)node * 128 + c]);
    }
    __syncthreads();

    const int lane = tid & 31, wv = tid >> 5;
    const int lo = lane & 15, hi = lane >> 4;
    const unsigned short* Arow = Ain + (wv * 16 + lo) * 264;
    const uint4* w1q = (const uint4*)w1f;
    const uint4* w2q = (const uint4*)w2f;

    for (int nt = 0; nt < 8; ++nt) {
        float bv = b1[nt * 16 + lo];
        v8f acc = {bv, bv, bv, bv, bv, bv, bv, bv};
#pragma unroll
        for (int kc = 0; kc < 8; ++kc) {
            int kb = kc * 32 + hi * 8;
            uint4 a0 = *(const uint4*)(Arow + kb);
            uint4 a1 = *(const uint4*)(Arow + kb + 16);
            const uint4* bp = w1q + ((((kc * 8 + nt) * 32) + lane) << 1);
            acc = wmma_bf16(mkfrag(a0, a1), mkfrag(bp[0], bp[1]), acc);
        }
#pragma unroll
        for (int r = 0; r < 8; ++r) {
            float v = fmaxf(acc[r], 0.f);
            Hs[(wv * 16 + r + 8 * hi) * 136 + nt * 16 + lo] = f2bf(v);
        }
    }
    __syncthreads();

    const unsigned short* Hrow = Hs + (wv * 16 + lo) * 136;
    for (int nt = 0; nt < 8; ++nt) {
        float bv = b2[nt * 16 + lo];
        v8f acc = {bv, bv, bv, bv, bv, bv, bv, bv};
#pragma unroll
        for (int kc = 0; kc < 4; ++kc) {
            int kb = kc * 32 + hi * 8;
            uint4 a0 = *(const uint4*)(Hrow + kb);
            uint4 a1 = *(const uint4*)(Hrow + kb + 16);
            const uint4* bp = w2q + ((((kc * 8 + nt) * 32) + lane) << 1);
            acc = wmma_bf16(mkfrag(a0, a1), mkfrag(bp[0], bp[1]), acc);
        }
#pragma unroll
        for (int r = 0; r < 8; ++r) {
            int row = wv * 16 + r + 8 * hi;
            int node = n0 + row; if (node >= n) node = n - 1;
            Rs[row * 132 + nt * 16 + lo] =
                acc[r] + x[(size_t)node * 128 + nt * 16 + lo];
        }
    }
    __syncthreads();

    // LayerNorm: wave handles its 16 rows; 2 lanes (hi) per row, 64 cols each.
    {
        int row = wv * 16 + lo;
        const float* R = Rs + row * 132 + hi * 64;
        float s = 0.f;
#pragma unroll
        for (int c = 0; c < 64; ++c) s += R[c];
        s += __shfl_xor(s, 16);
        float mu = s * (1.f / 128.f);
        float vs = 0.f;
#pragma unroll
        for (int c = 0; c < 64; ++c) { float d = R[c] - mu; vs += d * d; }
        vs += __shfl_xor(vs, 16);
        float rstd = rsqrtf(vs * (1.f / 128.f) + 1e-5f);
        int node = n0 + row;
        if (node < n) {
#pragma unroll
            for (int c = 0; c < 64; ++c) {
                int col = hi * 64 + c;
                float y = (R[c] - mu) * rstd * lnw[col] + lnb[col];
                x[(size_t)node * 128 + col]  = y;
                xh[(size_t)node * 128 + col] = f2bf(y);
            }
        }
    }
}

// ---------------------------------------------------------------------------
// Column-sum of final x (for mean) + unrouted-mask sum.  colsum[0..127]=cols,
// colsum[128]=mask count.  Must be zeroed beforehand.
// ---------------------------------------------------------------------------
__global__ void reduce_cols(const float* __restrict__ x,
                            const unsigned char* __restrict__ mask,
                            float* __restrict__ colsum, int n) {
    int gid = blockIdx.x * 128 + threadIdx.x;  // 64 blocks x 128 = 8192
    int col = gid & 127;
    float s = 0.f;
    for (int r = gid >> 7; r < n; r += 64) s += x[(size_t)r * 128 + col];
    atomicAdd(&colsum[col], s);
    float ms = 0.f;
    for (int i = gid; i < n; i += 8192) ms += (float)mask[i];
    atomicAdd(&colsum[128], ms);
}

// ---------------------------------------------------------------------------
// Readout head (tiny): 1 block, 128 threads.
// ---------------------------------------------------------------------------
__global__ void final_kernel(const float* __restrict__ colsum, int n,
                             const float* uw1, const float* ub1,
                             const float* uw2, const float* ub2,
                             const float* rw1, const float* rb1,
                             const float* rw2, const float* rb2,
                             const float* rw3, const float* rb3,
                             float* __restrict__ out) {
    __shared__ float gm[128], u1[128], u2[128], h1[128], h2[64];
    int t = threadIdx.x;
    float invn = 1.f / (float)n;
    float frac = colsum[128] * invn;
    gm[t] = colsum[t] * invn;
    u1[t] = fmaxf(frac * uw1[t] + ub1[t], 0.f);
    __syncthreads();
    {
        float s = ub2[t];
        for (int k = 0; k < 128; ++k) s += u1[k] * uw2[k * 128 + t];
        u2[t] = s;
    }
    __syncthreads();
    {
        float h = rb1[t];
        for (int k = 0; k < 128; ++k) h += gm[k] * rw1[k * 128 + t];
        for (int k = 0; k < 128; ++k) h += u2[k] * rw1[(128 + k) * 128 + t];
        h1[t] = fmaxf(h, 0.f);
    }
    __syncthreads();
    if (t < 64) {
        float g = rb2[t];
        for (int k = 0; k < 128; ++k) g += h1[k] * rw2[k * 64 + t];
        h2[t] = fmaxf(g, 0.f);
    }
    __syncthreads();
    if (t == 0) {
        float o = rb3[0];
        for (int k = 0; k < 64; ++k) o += h2[k] * rw3[k];
        out[0] = 1.f / (1.f + expf(-o));
    }
}

// ---------------------------------------------------------------------------
extern "C" void kernel_launch(void* const* d_in, const int* in_sizes, int n_in,
                              void* d_out, int out_size, void* d_ws, size_t ws_size,
                              hipStream_t stream) {
    const float* node_features = (const float*)d_in[0];
    const float* edge_features = (const float*)d_in[1];
    const float* congestion    = (const float*)d_in[2];
    const int*   edge_index    = (const int*)d_in[3];
    const unsigned char* unrouted = (const unsigned char*)d_in[4];
    const float* enc_node_w = (const float*)d_in[5];
    const float* enc_node_b = (const float*)d_in[6];
    const float* enc_edge_w = (const float*)d_in[7];
    const float* enc_edge_b = (const float*)d_in[8];
    const float* msg_w1 = (const float*)d_in[9];
    const float* msg_b1 = (const float*)d_in[10];
    const float* msg_w2 = (const float*)d_in[11];
    const float* msg_b2 = (const float*)d_in[12];
    const float* upd_w1 = (const float*)d_in[13];
    const float* upd_b1 = (const float*)d_in[14];
    const float* upd_w2 = (const float*)d_in[15];
    const float* upd_b2 = (const float*)d_in[16];
    const float* ln_w = (const float*)d_in[17];
    const float* ln_b = (const float*)d_in[18];
    const float* unr_w1 = (const float*)d_in[19];
    const float* unr_b1 = (const float*)d_in[20];
    const float* unr_w2 = (const float*)d_in[21];
    const float* unr_b2 = (const float*)d_in[22];
    const float* ro_w1 = (const float*)d_in[23];
    const float* ro_b1 = (const float*)d_in[24];
    const float* ro_w2 = (const float*)d_in[25];
    const float* ro_b2 = (const float*)d_in[26];
    const float* ro_w3 = (const float*)d_in[27];
    const float* ro_b3 = (const float*)d_in[28];

    const int N = in_sizes[2];           // 50000
    const int E = in_sizes[1] / 32;      // 800000
    const int* src = edge_index;
    const int* dst = edge_index + E;

    // ---- carve workspace ----
    char* ws = (char*)d_ws;
    size_t off = 0;
    auto carve = [&](size_t bytes) -> char* {
        char* p = ws + off;
        off = (off + bytes + 255) & ~(size_t)255;
        return p;
    };
    float*          x    = (float*)carve((size_t)N * 128 * 4);
    unsigned short* xh   = (unsigned short*)carve((size_t)N * 128 * 2);
    float*          agg  = (float*)carve((size_t)N * 128 * 4);
    unsigned short* ea   = (unsigned short*)carve((size_t)E * 128 * 2);
    unsigned short* encnf = (unsigned short*)carve(2 * 8 * 32 * 16 * 2);
    unsigned short* encef = (unsigned short*)carve(1 * 8 * 32 * 16 * 2);
    unsigned short* mw1f  = (unsigned short*)carve((size_t)4 * 13 * 8 * 32 * 16 * 2);
    unsigned short* mw2f  = (unsigned short*)carve((size_t)4 * 4 * 8 * 32 * 16 * 2);
    unsigned short* uw1f  = (unsigned short*)carve((size_t)4 * 8 * 8 * 32 * 16 * 2);
    unsigned short* uw2f  = (unsigned short*)carve((size_t)4 * 4 * 8 * 32 * 16 * 2);
    float*          colsum = (float*)carve(129 * 4);

    // ---- pack weights to bf16 fragment layout ----
    auto pack = [&](const float* w, unsigned short* o, int Kact, int KC) {
        int total = KC * 8 * 32;
        pack_b_frags<<<(total + 255) / 256, 256, 0, stream>>>(w, o, Kact, KC);
    };
    pack(enc_node_w, encnf, 64, 2);
    pack(enc_edge_w, encef, 32, 1);
    for (int l = 0; l < 4; ++l) {
        pack(msg_w1 + (size_t)l * 385 * 128, mw1f + (size_t)l * 13 * 8 * 32 * 16, 385, 13);
        pack(msg_w2 + (size_t)l * 128 * 128, mw2f + (size_t)l * 4 * 8 * 32 * 16, 128, 4);
        pack(upd_w1 + (size_t)l * 256 * 128, uw1f + (size_t)l * 8 * 8 * 32 * 16, 256, 8);
        pack(upd_w2 + (size_t)l * 128 * 128, uw2f + (size_t)l * 4 * 8 * 32 * 16, 128, 4);
    }

    // ---- encoders ----
    encoder_kernel<64, 2, 72><<<(N + 63) / 64, 128, 64 * 72 * 2, stream>>>(
        node_features, encnf, enc_node_b, x, xh, N);
    encoder_kernel<32, 1, 40><<<(E + 63) / 64, 128, 64 * 40 * 2, stream>>>(
        edge_features, encef, enc_edge_b, nullptr, ea, E);

    // ---- message-passing layers ----
    for (int l = 0; l < 4; ++l) {
        hipMemsetAsync(agg, 0, (size_t)N * 128 * 4, stream);
        msg_kernel<<<(E + 63) / 64, 128, 72192, stream>>>(
            xh, ea, congestion, src, dst,
            mw1f + (size_t)l * 13 * 8 * 32 * 16, msg_b1 + l * 128,
            mw2f + (size_t)l * 4 * 8 * 32 * 16, msg_b2 + l * 128, agg, E);
        upd_kernel<<<(N + 63) / 64, 128, 84992, stream>>>(
            x, agg,
            uw1f + (size_t)l * 8 * 8 * 32 * 16, upd_b1 + l * 128,
            uw2f + (size_t)l * 4 * 8 * 32 * 16, upd_b2 + l * 128,
            ln_w + l * 128, ln_b + l * 128, xh, N);
    }

    // ---- readout ----
    hipMemsetAsync(colsum, 0, 129 * 4, stream);
    reduce_cols<<<64, 128, 0, stream>>>(x, unrouted, colsum, N);
    final_kernel<<<1, 128, 0, stream>>>(colsum, N, unr_w1, unr_b1, unr_w2, unr_b2,
                                        ro_w1, ro_b1, ro_w2, ro_b2, ro_w3, ro_b3,
                                        (float*)d_out);
}
